// AttentionHead_61314953118282
// MI455X (gfx1250) — matrix operand order repaired and checked by reading, not compiled, not run on hardware
//
#include <hip/hip_runtime.h>
#include <hip/hip_bf16.h>

typedef __bf16 bf16;
typedef __attribute__((ext_vector_type(16))) __bf16 v16bf;
typedef __attribute__((ext_vector_type(8)))  __bf16 v8bf;
typedef __attribute__((ext_vector_type(8)))  float  v8f;

#define B_   512
#define T_   80
#define C_   512
#define H_   256
#define V_   6625
#define S_   25
#define G3H  768      // 3*H
#define CV   7137     // C + V (gru_wih inner dim)
#define VP   6656     // V padded to multiple of 64 (N register-blocking factor)

// ---------------- conversion kernels (one-time prep) ----------------

__global__ void f32_to_bf16_kernel(const float* __restrict__ src, bf16* __restrict__ dst, size_t n) {
  size_t i = (size_t)blockIdx.x * blockDim.x + threadIdx.x;
  size_t stride = (size_t)gridDim.x * blockDim.x;
  for (; i < n; i += stride) dst[i] = (bf16)src[i];
}

// extract gru_wih[:, 0:C] (K-contiguous slice of the [3H, C+V] matrix) as bf16 [3H, C]
__global__ void wih_slice_bf16_kernel(const float* __restrict__ wih, bf16* __restrict__ dst) {
  size_t i = (size_t)blockIdx.x * blockDim.x + threadIdx.x;
  if (i >= (size_t)G3H * C_) return;
  size_t j = i >> 9;           // C_ = 512
  size_t c = i & 511;
  dst[i] = (bf16)wih[j * (size_t)CV + c];
}

// gen_w [V,H] -> bf16 [VP,H], zero-padded rows V..VP-1
__global__ void genw_pad_bf16_kernel(const float* __restrict__ gw, bf16* __restrict__ dst) {
  size_t i = (size_t)blockIdx.x * blockDim.x + threadIdx.x;
  if (i >= (size_t)VP * H_) return;
  size_t v = i >> 8;           // H_ = 256
  size_t h = i & 255;
  dst[i] = (v < (size_t)V_) ? (bf16)gw[v * (size_t)H_ + h] : (bf16)0.0f;
}

// ---------------- bf16 WMMA GEMM: C = A[M,K] * W[Npad,K]^T (+bias) ----------------
// One wave per 16x64 output tile (4 N-subtiles): A fragment loaded once per K-step
// and reused across 4 v_wmma_f32_16x16x32_bf16. K mult of 32; M mult of 16; Npad mult of 64.
// Stores only columns n < Nreal, with leading dim ldc.

__global__ void wmma_gemm_bf16_kernel(const bf16* __restrict__ A, const bf16* __restrict__ W,
                                      const float* __restrict__ bias, float* __restrict__ Cmat,
                                      int M, int Npad, int K, int Nreal, int ldc) {
  int wave    = blockIdx.x * (blockDim.x >> 5) + (threadIdx.x >> 5);
  int tilesN4 = Npad >> 6;                // 64-wide N tiles
  int tilesM  = M >> 4;
  if (wave >= tilesM * tilesN4) return;   // wave-uniform branch: EXEC stays all-1s for WMMA

  int tm  = wave / tilesN4;
  int tn4 = wave % tilesN4;
  int lane = threadIdx.x & 31;
  int l15  = lane & 15;
  int hs   = lane >> 4;                   // half-select: lanes 0-15 vs 16-31

  const bf16* Arow  = A + (size_t)(tm * 16 + l15) * K;        // activation row
  const bf16* Wbase = W + (size_t)(tn4 * 64 + l15) * K;       // first of 4 weight sub-tiles
  size_t wstep = (size_t)16 * K;                              // row stride between sub-tiles

  v8f acc0 = {}, acc1 = {}, acc2 = {}, acc3 = {};
  for (int k = 0; k < K; k += 32) {
    // A 16x32 layout: lanes 0-15 hold K 0..15, lanes 16-31 hold K 16..31, contiguous per lane
    v16bf av = *(const v16bf*)(Arow + k + hs * 16);
    // B 32x16 layout: lanes 0-15 hold K 0..15, lanes 16-31 hold K 16..31, contiguous per lane
    const bf16* wk = Wbase + k + hs * 16;
    v16bf b0 = *(const v16bf*)(wk);
    v16bf b1 = *(const v16bf*)(wk + wstep);
    v16bf b2 = *(const v16bf*)(wk + 2 * wstep);
    v16bf b3 = *(const v16bf*)(wk + 3 * wstep);
    if (k + 64 < K) {                     // gfx1250 global_prefetch_b8 two iterations ahead
      __builtin_prefetch(Arow + k + 64, 0, 1);
      __builtin_prefetch(Wbase + k + 64, 0, 1);
    }
    acc0 = __builtin_amdgcn_wmma_f32_16x16x32_bf16(false, av, false, b0, (short)0, acc0, false, false);
    acc1 = __builtin_amdgcn_wmma_f32_16x16x32_bf16(false, av, false, b1, (short)0, acc1, false, false);
    acc2 = __builtin_amdgcn_wmma_f32_16x16x32_bf16(false, av, false, b2, (short)0, acc2, false, false);
    acc3 = __builtin_amdgcn_wmma_f32_16x16x32_bf16(false, av, false, b3, (short)0, acc3, false, false);
  }

  // D layout: lane<16 -> rows tm*16+0..7, col l15; lane>=16 -> rows tm*16+8..15
  int r0 = tm * 16 + hs * 8;
  v8f accs[4] = {acc0, acc1, acc2, acc3};
#pragma unroll
  for (int j = 0; j < 4; ++j) {
    int n = tn4 * 64 + j * 16 + l15;
    if (n < Nreal) {
      float badd = bias ? bias[n] : 0.0f;
#pragma unroll
      for (int i = 0; i < 8; ++i)
        Cmat[(size_t)(r0 + i) * ldc + n] = accs[j][i] + badd;
    }
  }
}

// ---------------- fused attention step: scores + softmax + context ----------------
// One block (256 threads = 8 waves) per batch element. proj/inputs stream from L2
// (proj 42 MB + inputs 84 MB + weights fit in the 192 MB global L2).

__global__ void attention_kernel(const float* __restrict__ proj,    // [B,T,H]
                                 const float* __restrict__ ph,      // [B,H]
                                 const float* __restrict__ inputs,  // [B,T,C]
                                 const float* __restrict__ score_w, // [H]
                                 bf16* __restrict__ context_bf) {   // [B,C]
  __shared__ float s_ph[H_];
  __shared__ float s_sw[H_];
  __shared__ float s_e[T_];
  int b = blockIdx.x, tid = threadIdx.x;
  s_ph[tid] = ph[(size_t)b * H_ + tid];
  s_sw[tid] = score_w[tid];
  __syncthreads();

  int wave = tid >> 5, lane = tid & 31;
  for (int t0 = 0; t0 < T_; t0 += 8) {       // 8 waves cover 8 timesteps per pass
    int t = t0 + wave;
    const float* pr = proj + ((size_t)b * T_ + t) * H_;
    float p = 0.0f;
#pragma unroll
    for (int h = lane; h < H_; h += 32)
      p += s_sw[h] * tanhf(pr[h] + s_ph[h]);
#pragma unroll
    for (int off = 16; off > 0; off >>= 1)
      p += __shfl_xor(p, off, 32);
    if (lane == 0) s_e[t] = p;
  }
  __syncthreads();

  // softmax over T=80 (each thread redundantly reduces; cheap and barrier-free)
  float mx = -3.0e38f;
  for (int t = 0; t < T_; ++t) mx = fmaxf(mx, s_e[t]);
  float sum = 0.0f;
  for (int t = 0; t < T_; ++t) sum += __expf(s_e[t] - mx);
  float inv = 1.0f / sum;
  __syncthreads();
  if (tid < T_) s_e[tid] = __expf(s_e[tid] - mx) * inv;
  __syncthreads();

  // context[b,c] = sum_t alpha[t] * inputs[b,t,c]; thread owns c = tid, tid+256
  float acc0 = 0.0f, acc1 = 0.0f;
  const float* inb = inputs + (size_t)b * T_ * C_;
  for (int t = 0; t < T_; ++t) {
    float a = s_e[t];
    acc0 += a * inb[(size_t)t * C_ + tid];
    acc1 += a * inb[(size_t)t * C_ + tid + 256];
  }
  context_bf[(size_t)b * C_ + tid]       = (bf16)acc0;
  context_bf[(size_t)b * C_ + tid + 256] = (bf16)acc1;
}

// ---------------- fused GRU gate math + one-hot column gather ----------------
// gi already holds context@wih[:, :C]^T + bih ; gh holds h@whh^T + bhh.
// One-hot concat contributes column C+tgt of gru_wih (gathered from f32 weights).

__global__ void gru_gate_kernel(const float* __restrict__ gi, const float* __restrict__ gh,
                                const int* __restrict__ targets, int step,
                                const float* __restrict__ gru_wih,
                                float* __restrict__ hidden, bf16* __restrict__ hidden_bf,
                                float* __restrict__ hiddens) {
  int b = blockIdx.x, j = threadIdx.x;
  int tgt = targets[b * S_ + step];
  size_t col = (size_t)C_ + (size_t)tgt;
  const float* gib = gi + (size_t)b * G3H;
  const float* ghb = gh + (size_t)b * G3H;
  float i_r = gib[j]            + gru_wih[(size_t)j * CV + col];
  float i_z = gib[H_ + j]       + gru_wih[((size_t)H_ + j) * CV + col];
  float i_n = gib[2 * H_ + j]   + gru_wih[((size_t)2 * H_ + j) * CV + col];
  float h_r = ghb[j];
  float h_z = ghb[H_ + j];
  float h_n = ghb[2 * H_ + j];
  float r = 1.0f / (1.0f + __expf(-(i_r + h_r)));
  float z = 1.0f / (1.0f + __expf(-(i_z + h_z)));
  float n = tanhf(i_n + r * h_n);
  float h = hidden[(size_t)b * H_ + j];
  float nh = (1.0f - z) * n + z * h;
  hidden[(size_t)b * H_ + j]    = nh;
  hidden_bf[(size_t)b * H_ + j] = (bf16)nh;
  hiddens[((size_t)b * S_ + step) * H_ + j] = nh;   // [B,S,H] rows for generator GEMM
}

// ---------------- host orchestration ----------------

extern "C" void kernel_launch(void* const* d_in, const int* in_sizes, int n_in,
                              void* d_out, int out_size, void* d_ws, size_t ws_size,
                              hipStream_t stream) {
  const float* inputs  = (const float*)d_in[0];
  const int*   targets = (const int*)d_in[1];
  // d_in[2] = batch_max_length scalar (compile-time S_ = 25)
  const float* i2h_w   = (const float*)d_in[3];
  const float* h2h_w   = (const float*)d_in[4];
  const float* h2h_b   = (const float*)d_in[5];
  const float* score_w = (const float*)d_in[6];
  const float* gru_wih = (const float*)d_in[7];
  const float* gru_whh = (const float*)d_in[8];
  const float* gru_bih = (const float*)d_in[9];
  const float* gru_bhh = (const float*)d_in[10];
  const float* gen_w   = (const float*)d_in[11];
  const float* gen_b   = (const float*)d_in[12];
  float* out = (float*)d_out;

  char* p = (char*)d_ws;
  auto carve = [&](size_t bytes) -> void* {
    void* q = (void*)p;
    p += (bytes + 255) & ~(size_t)255;
    return q;
  };
  bf16*  inputs_bf  = (bf16*) carve((size_t)B_ * T_ * C_ * sizeof(bf16));
  bf16*  i2h_bf     = (bf16*) carve((size_t)H_ * C_ * sizeof(bf16));
  bf16*  h2h_bf     = (bf16*) carve((size_t)H_ * H_ * sizeof(bf16));
  bf16*  whh_bf     = (bf16*) carve((size_t)G3H * H_ * sizeof(bf16));
  bf16*  wih_bf     = (bf16*) carve((size_t)G3H * C_ * sizeof(bf16));
  bf16*  genw_bf    = (bf16*) carve((size_t)VP * H_ * sizeof(bf16));
  float* proj       = (float*)carve((size_t)B_ * T_ * H_ * sizeof(float));
  float* ph         = (float*)carve((size_t)B_ * H_ * sizeof(float));
  bf16*  ctx_bf     = (bf16*) carve((size_t)B_ * C_ * sizeof(bf16));
  float* gi         = (float*)carve((size_t)B_ * G3H * sizeof(float));
  float* gh         = (float*)carve((size_t)B_ * G3H * sizeof(float));
  float* hidden     = (float*)carve((size_t)B_ * H_ * sizeof(float));
  bf16*  hidden_bf  = (bf16*) carve((size_t)B_ * H_ * sizeof(bf16));
  float* hiddens    = (float*)carve((size_t)B_ * S_ * H_ * sizeof(float));
  bf16*  hiddens_bf = (bf16*) carve((size_t)B_ * S_ * H_ * sizeof(bf16));

  // one-time bf16 prep
  f32_to_bf16_kernel<<<4096, 256, 0, stream>>>(inputs, inputs_bf, (size_t)B_ * T_ * C_);
  f32_to_bf16_kernel<<<64,   256, 0, stream>>>(i2h_w, i2h_bf, (size_t)H_ * C_);
  f32_to_bf16_kernel<<<64,   256, 0, stream>>>(h2h_w, h2h_bf, (size_t)H_ * H_);
  f32_to_bf16_kernel<<<96,   256, 0, stream>>>(gru_whh, whh_bf, (size_t)G3H * H_);
  wih_slice_bf16_kernel<<<(G3H * C_ + 255) / 256, 256, 0, stream>>>(gru_wih, wih_bf);
  genw_pad_bf16_kernel<<<((int)(((size_t)VP * H_ + 255) / 256)), 256, 0, stream>>>(gen_w, genw_bf);

  hipMemsetAsync(hidden,    0, (size_t)B_ * H_ * sizeof(float), stream);
  hipMemsetAsync(hidden_bf, 0, (size_t)B_ * H_ * sizeof(bf16),  stream);

  auto gemm = [&](const bf16* A, const bf16* W, const float* bias, float* Cm,
                  int M, int Npad, int K, int Nreal, int ldc) {
    int tiles  = (M >> 4) * (Npad >> 6);              // 16x64 tiles per wave
    int blocks = (tiles + 7) / 8;                     // 8 waves (256 threads) per block
    wmma_gemm_bf16_kernel<<<blocks, 256, 0, stream>>>(A, W, bias, Cm, M, Npad, K, Nreal, ldc);
  };

  // step-invariant projection: proj[b,t,:] = inputs[b,t,:] @ i2h_w^T
  gemm(inputs_bf, i2h_bf, nullptr, proj, B_ * T_, H_, C_, H_, H_);

  for (int s = 0; s < S_; ++s) {
    gemm(hidden_bf, h2h_bf, h2h_b, ph, B_, H_, H_, H_, H_);                 // ph = h@h2h^T + b
    attention_kernel<<<B_, 256, 0, stream>>>(proj, ph, inputs, score_w, ctx_bf);
    gemm(ctx_bf, wih_bf, gru_bih, gi, B_, G3H, C_, G3H, G3H);               // gi = ctx@wih[:,:C]^T + bih
    gemm(hidden_bf, whh_bf, gru_bhh, gh, B_, G3H, H_, G3H, G3H);            // gh = h@whh^T + bhh
    gru_gate_kernel<<<B_, 256, 0, stream>>>(gi, gh, targets, s, gru_wih,
                                            hidden, hidden_bf, hiddens);
  }

  // generator: probs[b,s,:] = hiddens[b,s,:] @ gen_w^T + gen_b
  f32_to_bf16_kernel<<<1024, 256, 0, stream>>>(hiddens, hiddens_bf, (size_t)B_ * S_ * H_);
  gemm(hiddens_bf, genw_bf, gen_b, out, B_ * S_, VP, H_, V_, V_);
}
